// SA_CovaMNET_87806311399638
// MI455X (gfx1250) — matrix operand order repaired and checked by reading, not compiled, not run on hardware
//
#include <hip/hip_runtime.h>

typedef _Float16 half_t;
typedef __attribute__((ext_vector_type(16))) _Float16 v16h;
typedef __attribute__((ext_vector_type(8)))  _Float16 v8h;
typedef __attribute__((ext_vector_type(8)))  float    v8f;

#define NIMG 178   // 128 query + 5*10 support

// ---------------------------------------------------------------- WMMA helpers
__device__ __forceinline__ v8f wmma16(v16h a, v16h b, v8f c) {
  return __builtin_amdgcn_wmma_f32_16x16x32_f16(false, a, false, b, (short)0, c,
                                                false, false);
}

// A fragment, K contiguous in memory. p = rowBase + k0 + kbA (kbA = laneHi?8:0).
// elements [0..7] = p[0..7], [8..15] = p[16..23]; p must be 16B aligned.
__device__ __forceinline__ v16h frag_kcontig(const half_t* p) {
  v8h lo = *(const v8h*)(p);
  v8h hi = *(const v8h*)(p + 16);
  v16h r;
#pragma unroll
  for (int i = 0; i < 8; ++i) { r[i] = lo[i]; r[8 + i] = hi[i]; }
  return r;
}

// A fragment, K strided by ld, upper-K half zero (K=16 padded to 32).
// p = base + kbA*ld + col.
__device__ __forceinline__ v16h frag_kstride_pad(const half_t* p, int ld) {
  v16h r;
#pragma unroll
  for (int i = 0; i < 8; ++i) { r[i] = p[i * ld]; r[8 + i] = (half_t)0.f; }
  return r;
}

// B fragment, K contiguous. p = rowBase + k0 + koB (koB = laneHi?8:0); [0..7]=p[0..7], [8..15]=p[16..23].
__device__ __forceinline__ v16h fragB_kcontig(const half_t* p) {
  v8h lo = *(const v8h*)(p);
  v8h hi = *(const v8h*)(p + 16);
  v16h r;
#pragma unroll
  for (int i = 0; i < 8; ++i) { r[i] = lo[i]; r[8 + i] = hi[i]; }
  return r;
}

// B fragment for K=16 padded to 32: elements [8..15] zero; upper lane half takes k=8..15.
__device__ __forceinline__ v16h fragB_pad(const half_t* p, int ld, bool zero) {
  v16h r;
#pragma unroll
  for (int e = 0; e < 16; ++e) r[e] = e >= 8 ? (half_t)0.f : p[(e + (zero ? 8 : 0)) * ld];
  return r;
}

__device__ __forceinline__ float leaky(float v)    { return v > 0.f ? v : 0.2f * v; }
__device__ __forceinline__ float sigmoidf_(float v){ return 1.f / (1.f + expf(-v)); }
__device__ __forceinline__ int   group_of(int img) { return img < 128 ? 0 : 1 + (img - 128) / 10; }

__device__ __forceinline__ void block_reduce2(float& s, float& q, float* bs, float* bq) {
  int tid = threadIdx.x;
  bs[tid] = s; bq[tid] = q;
  __syncthreads();
  for (int k = 128; k > 0; k >>= 1) {
    if (tid < k) { bs[tid] += bs[tid + k]; bq[tid] += bq[tid + k]; }
    __syncthreads();
  }
  s = bs[0]; q = bq[0];
  __syncthreads();
}

// ---------------------------------------------------------------- prep
// repack conv weights [o][i][3][3] -> f16 [tap][o][i]; q/k/v weights -> f16
__global__ void __launch_bounds__(256) k_prep(
    const float* fW1, const float* fW2, const float* fW3,
    const float* Wq, const float* Wk, const float* Wv,
    half_t* Wt1, half_t* Wt2, half_t* Wt3,
    half_t* Wqh, half_t* Wkh, half_t* Wvh) {
  int id = blockIdx.x * 256 + threadIdx.x;
  if (id < 36864) {                       // 64*64*9
    int o = id / 576, rem = id % 576, i = rem / 9, t = rem % 9;
    int dst = t * 4096 + o * 64 + i;
    Wt1[dst] = (half_t)fW1[id];
    Wt2[dst] = (half_t)fW2[id];
    Wt3[dst] = (half_t)fW3[id];
  }
  if (id < 4096) {
    Wqh[id] = (half_t)Wq[id];
    Wkh[id] = (half_t)Wk[id];
    Wvh[id] = (half_t)Wv[id];
  }
}

// ---------------------------------------------------------------- conv0 (1ch)
__global__ void __launch_bounds__(256) k_conv0_stats(
    const float* in1, const float* in2, const float* fW0, float* part) {
  int img = blockIdx.x >> 6, ch = blockIdx.x & 63;
  const float* X = (img < 128) ? in1 + img * 4096 : in2 + (img - 128) * 4096;
  float w[9];
#pragma unroll
  for (int t = 0; t < 9; ++t) w[t] = fW0[ch * 9 + t];
  float s = 0.f, q = 0.f;
  for (int i = threadIdx.x; i < 4096; i += 256) {
    int py = i >> 6, px = i & 63;
    float acc = 0.f;
#pragma unroll
    for (int t = 0; t < 9; ++t) {
      int sy = py + t / 3 - 1, sx = px + t % 3 - 1;
      bool inb = (sy >= 0) && (sy < 64) && (sx >= 0) && (sx < 64);
      int cy = sy < 0 ? 0 : (sy > 63 ? 63 : sy);
      int cx = sx < 0 ? 0 : (sx > 63 ? 63 : sx);
      acc += (inb ? w[t] : 0.f) * X[cy * 64 + cx];   // branchless clamped load
    }
    s += acc; q += acc * acc;
  }
  __shared__ float bs[256], bq[256];
  block_reduce2(s, q, bs, bq);
  if (threadIdx.x == 0) {
    part[(img * 64 + ch) * 2]     = s;
    part[(img * 64 + ch) * 2 + 1] = q;
  }
}

// deterministic per-(group,channel) finalize: mean + 1/sqrt(var+eps)
__global__ void k_finalize_stats(const float* part, float* mi, int HW) {
  int id = threadIdx.x;
  if (id >= 384) return;
  int g = id >> 6, ch = id & 63;
  int start = (g == 0) ? 0 : 128 + (g - 1) * 10;
  int cnt   = (g == 0) ? 128 : 10;
  float s = 0.f, q = 0.f;
  for (int k = 0; k < cnt; ++k) {
    s += part[((start + k) * 64 + ch) * 2];
    q += part[((start + k) * 64 + ch) * 2 + 1];
  }
  float n = (float)cnt * (float)HW;
  float m = s / n;
  float v = q / n - m * m;
  mi[id * 2]     = m;
  mi[id * 2 + 1] = rsqrtf(v + 1e-5f);
}

// conv0 + BN + leaky + 2x2 maxpool, output NHWC [img][32*32][64] f16
__global__ void __launch_bounds__(256) k_conv0_apply(
    const float* in1, const float* in2, const float* fW0,
    const float* mi, const float* g0, const float* b0, half_t* x1) {
  int id = blockIdx.x * 256 + threadIdx.x;        // img*1024*64 + p*64 + ch
  int ch = id & 63, p = (id >> 6) & 1023, img = id >> 16;
  int oy = p >> 5, ox = p & 31;
  const float* X = (img < 128) ? in1 + img * 4096 : in2 + (img - 128) * 4096;
  int grp = group_of(img);
  float mean = mi[(grp * 64 + ch) * 2], inv = mi[(grp * 64 + ch) * 2 + 1];
  float gg = g0[ch], bb = b0[ch];
  float w[9];
#pragma unroll
  for (int t = 0; t < 9; ++t) w[t] = fW0[ch * 9 + t];
  float best = -1e30f;
#pragma unroll
  for (int sub = 0; sub < 4; ++sub) {
    int py = 2 * oy + (sub >> 1), px = 2 * ox + (sub & 1);
    float acc = 0.f;
#pragma unroll
    for (int t = 0; t < 9; ++t) {
      int sy = py + t / 3 - 1, sx = px + t % 3 - 1;
      bool inb = (sy >= 0) && (sy < 64) && (sx >= 0) && (sx < 64);
      int cy = sy < 0 ? 0 : (sy > 63 ? 63 : sy);
      int cx = sx < 0 ? 0 : (sx > 63 ? 63 : sx);
      acc += (inb ? w[t] : 0.f) * X[cy * 64 + cx];
    }
    best = fmaxf(best, leaky((acc - mean) * inv * gg + bb));
  }
  x1[id] = (half_t)best;
}

// ------------------------------------------------------- conv1..3 implicit GEMM
// NHWC activations; weights staged in LDS; per wave 64(Cout) x 16(pixels) tile.
template <int H, int W>
__global__ void __launch_bounds__(256) k_conv_gemm(
    const half_t* xin, const half_t* Wt, half_t* yout) {
  constexpr int HW = H * W;
  constexpr int TILES = HW / 128;
  __shared__ __attribute__((aligned(16))) half_t lw[36864];   // [tap][o][i] 72KB
  {
    const v8h* src = (const v8h*)Wt;
    v8h* dst = (v8h*)lw;
    for (int i = threadIdx.x; i < 4608; i += 256) dst[i] = src[i];
  }
  __syncthreads();
  int img = blockIdx.x / TILES, tile = blockIdx.x % TILES;
  int lane = threadIdx.x & 31, wv = threadIdx.x >> 5;
  bool hiL = (lane & 16) != 0;
  int kbA = hiL ? 8 : 0, koB = hiL ? 8 : 0;
  int p = tile * 128 + wv * 16 + (lane & 15);
  int py = p / W, px = p % W;
  const half_t* X = xin + img * (HW * 64);
  v8f acc0 = {}, acc1 = {}, acc2 = {}, acc3 = {};
#pragma unroll
  for (int t = 0; t < 9; ++t) {
    int sy = py + t / 3 - 1, sx = px + (t % 3) - 1;
    bool ok = (sy >= 0) && (sy < H) && (sx >= 0) && (sx < W);
    int sidx = ok ? (sy * W + sx) : 0;                 // clamped, always valid
    const half_t* Xp = X + sidx * 64;
    __builtin_prefetch(Xp, 0, 1);
    const half_t* wt = lw + t * 4096;
#pragma unroll
    for (int kc = 0; kc < 2; ++kc) {
      v16h braw = fragB_kcontig(Xp + kc * 32 + koB);   // 2x16B contiguous loads
      v16h b;
#pragma unroll
      for (int e = 0; e < 16; ++e) b[e] = ok ? braw[e] : (half_t)0.f;  // cndmask
      const half_t* wrow = wt + kc * 32 + kbA;
      v16h a0 = frag_kcontig(wrow + ((lane & 15) + 0)  * 64);
      v16h a1 = frag_kcontig(wrow + ((lane & 15) + 16) * 64);
      v16h a2 = frag_kcontig(wrow + ((lane & 15) + 32) * 64);
      v16h a3 = frag_kcontig(wrow + ((lane & 15) + 48) * 64);
      acc0 = wmma16(a0, b, acc0);
      acc1 = wmma16(a1, b, acc1);
      acc2 = wmma16(a2, b, acc2);
      acc3 = wmma16(a3, b, acc3);
    }
  }
  half_t* Y = yout + img * (HW * 64) + p * 64;
  int rbase = hiL ? 8 : 0;
  v8h o0, o1, o2, o3;
#pragma unroll
  for (int r = 0; r < 8; ++r) {
    o0[r] = (half_t)acc0[r]; o1[r] = (half_t)acc1[r];
    o2[r] = (half_t)acc2[r]; o3[r] = (half_t)acc3[r];
  }
  *(v8h*)(Y + 0  + rbase) = o0;
  *(v8h*)(Y + 16 + rbase) = o1;
  *(v8h*)(Y + 32 + rbase) = o2;
  *(v8h*)(Y + 48 + rbase) = o3;
}

__global__ void __launch_bounds__(256) k_bn_stats(const half_t* y, float* part, int HW) {
  int img = blockIdx.x >> 6, ch = blockIdx.x & 63;
  const half_t* Y = y + img * (HW * 64) + ch;          // NHWC: stride 64 over pixels
  float s = 0.f, q = 0.f;
  for (int i = threadIdx.x; i < HW; i += 256) { float v = (float)Y[i * 64]; s += v; q += v * v; }
  __shared__ float bs[256], bq[256];
  block_reduce2(s, q, bs, bq);
  if (threadIdx.x == 0) {
    part[(img * 64 + ch) * 2]     = s;
    part[(img * 64 + ch) * 2 + 1] = q;
  }
}

__global__ void __launch_bounds__(256) k_bn_apply(
    const half_t* y, half_t* xo, const float* mi,
    const float* gg, const float* bb, int H, int W, int pool) {
  int id = blockIdx.x * 256 + threadIdx.x;
  int oH = pool ? H / 2 : H, oW = pool ? W / 2 : W;
  int oHW = oH * oW;
  int ch = id & 63;
  int p = (id >> 6) % oHW;
  int img = (id >> 6) / oHW;
  int grp = group_of(img);
  float mean = mi[(grp * 64 + ch) * 2], inv = mi[(grp * 64 + ch) * 2 + 1];
  float g = gg[ch], b = bb[ch];
  const half_t* Y = y + img * (H * W * 64);
  float r;
  if (pool) {
    int oy = p / oW, ox = p % oW;
    r = -1e30f;
#pragma unroll
    for (int s = 0; s < 4; ++s) {
      float v = (float)Y[((2 * oy + (s >> 1)) * W + 2 * ox + (s & 1)) * 64 + ch];
      r = fmaxf(r, leaky((v - mean) * inv * g + b));
    }
  } else {
    r = leaky(((float)Y[p * 64 + ch] - mean) * inv * g + b);
  }
  xo[id] = (half_t)r;                                   // NHWC linear
}

// ---------------------------------------------------------------- MHSA
// one block per (img, head): WMMA q/k/v projections, strip-wise scores+softmax,
// WMMA attn @ v^T with deterministic per-wave partial combine.
__global__ void __launch_bounds__(256) k_mhsa(
    const half_t* ff, const half_t* Wqh, const half_t* Wkh, const half_t* Wvh,
    const float* bq, const float* bk, const float* bv,
    const float* rel_h, const float* rel_w, float* attn_out) {
  __shared__ __attribute__((aligned(16))) half_t qs[4096], ks[4096], vs[4096], rs[4096];
  __shared__ __attribute__((aligned(16))) float Sst[4096];
  __shared__ float opart[8 * 256];
  __shared__ float red[256];
  int img = blockIdx.x >> 2, h = blockIdx.x & 3;
  int tid = threadIdx.x, lane = tid & 31, wv = tid >> 5;
  bool hiL = (lane & 16) != 0;
  int kbA = hiL ? 8 : 0, koB = hiL ? 8 : 0;
  const half_t* X = ff + img * 16384;                   // NHWC [p][c]

  // relative-position table r[d][p], p = i*16+j
  for (int d = 0; d < 16; ++d) {
    int i = tid >> 4, j = tid & 15;
    rs[d * 256 + tid] = (half_t)(rel_h[(h * 16 + d) * 16 + j] +
                                 rel_w[(h * 16 + d) * 16 + i]);
  }

  // q,k,v projections: 3 mats x 16 pixel tiles, M=16 (head dims), K=64 (channels)
  // independent accumulators per K-chunk avoid WMMA->WMMA hazard NOPs
  for (int job = wv; job < 48; job += 8) {
    int mat = job >> 4, pt = job & 15;
    const half_t* Wm  = (mat == 0) ? Wqh : (mat == 1) ? Wkh : Wvh;
    const float* bias = (mat == 0) ? bq  : (mat == 1) ? bk  : bv;
    half_t* dst       = (mat == 0) ? qs  : (mat == 1) ? ks  : vs;
    int col = pt * 16 + (lane & 15);
    const half_t* wr = Wm + (h * 16 + (lane & 15)) * 64;
    v16h a0 = frag_kcontig(wr + kbA);
    v16h b0 = fragB_kcontig(X + col * 64 + koB);
    v16h a1 = frag_kcontig(wr + 32 + kbA);
    v16h b1 = fragB_kcontig(X + col * 64 + 32 + koB);
    v8f accA = {}, accB = {};
    accA = wmma16(a0, b0, accA);
    accB = wmma16(a1, b1, accB);
    int rbase = hiL ? 8 : 0;
#pragma unroll
    for (int r = 0; r < 8; ++r) {
      int d = r + rbase;
      dst[d * 256 + col] = (half_t)(accA[r] + accB[r] + bias[h * 16 + d]);
    }
  }
  __syncthreads();

  half_t* Ah = (half_t*)Sst;   // f16 attention weights reuse the f32 score buffer
  for (int strip = 0; strip < 16; ++strip) {
    int nrow = strip * 16 + (lane & 15);
    // scores S[n,m] = q^T k + r^T q  (d=16 zero-padded to K=32), two indep accs
    for (int mt = wv; mt < 16; mt += 8) {
      int mcol = mt * 16 + (lane & 15);
      v16h a1 = frag_kstride_pad(qs + kbA * 256 + nrow, 256);
      v16h b1 = fragB_pad(ks + mcol, 256, hiL);
      v16h a2 = frag_kstride_pad(rs + kbA * 256 + nrow, 256);
      v16h b2 = fragB_pad(qs + mcol, 256, hiL);
      v8f accA = {}, accB = {};
      accA = wmma16(a1, b1, accA);
      accB = wmma16(a2, b2, accB);
      int rbase = hiL ? 8 : 0;
#pragma unroll
      for (int r = 0; r < 8; ++r) Sst[(r + rbase) * 256 + mcol] = accA[r] + accB[r];
    }
    __syncthreads();
    // softmax over m for the 16 rows of this strip (16 threads per row)
    int row = tid >> 4, seg = tid & 15;
    float m = -1e30f;
    for (int i = 0; i < 16; ++i) m = fmaxf(m, Sst[row * 256 + seg * 16 + i]);
    red[row * 16 + seg] = m;
    __syncthreads();
    if (seg == 0) {
      float mm = red[row * 16];
      for (int i = 1; i < 16; ++i) mm = fmaxf(mm, red[row * 16 + i]);
      red[row * 16] = mm;
    }
    __syncthreads();
    float rowmax = red[row * 16];
    float ex[16];
    float s = 0.f;
#pragma unroll
    for (int i = 0; i < 16; ++i) {
      ex[i] = expf(Sst[row * 256 + seg * 16 + i] - rowmax);
      s += ex[i];
    }
    __syncthreads();
    red[row * 16 + seg] = s;
    __syncthreads();
    if (seg == 0) {
      float ss = 0.f;
      for (int i = 0; i < 16; ++i) ss += red[row * 16 + i];
      red[row * 16] = ss;
    }
    __syncthreads();
    float inv = 1.f / red[row * 16];
#pragma unroll
    for (int i = 0; i < 16; ++i)
      Ah[row * 256 + seg * 16 + i] = (half_t)(ex[i] * inv);
    __syncthreads();
    // out[n,d] = sum_m attn[n,m] v[d,m]; wave wv takes K-chunk wv (K=256 over 8 waves)
    {
      int kc = wv;
      v16h a = frag_kcontig(Ah + (lane & 15) * 256 + kc * 32 + kbA);
      v16h b = fragB_kcontig(vs + (lane & 15) * 256 + kc * 32 + koB);
      v8f acc = {};
      acc = wmma16(a, b, acc);
      int rbase = hiL ? 8 : 0;
#pragma unroll
      for (int r = 0; r < 8; ++r)
        opart[wv * 256 + (r + rbase) * 16 + (lane & 15)] = acc[r];
    }
    __syncthreads();
    {
      float sum = 0.f;
#pragma unroll
      for (int w2 = 0; w2 < 8; ++w2) sum += opart[w2 * 256 + tid];
      int n = tid >> 4, d = tid & 15;
      attn_out[(img * 64 + h * 16 + d) * 256 + strip * 16 + n] = sum;   // CHW f32
    }
    __syncthreads();
  }
}

// LayerNorm over [C,W,H] + residual:  lnf = LN(attn)*g + b + f   (CHW f32)
__global__ void __launch_bounds__(256) k_ln(
    const float* attn, const half_t* ff, const float* lng, const float* lnb,
    float* lnf) {
  int img = blockIdx.x;
  const float* A = attn + img * 16384;
  float s = 0.f, q = 0.f;
  for (int i = threadIdx.x; i < 16384; i += 256) { float v = A[i]; s += v; q += v * v; }
  __shared__ float bs[256], bq[256];
  block_reduce2(s, q, bs, bq);
  float m = s / 16384.f;
  float inv = rsqrtf(q / 16384.f - m * m + 1e-5f);
  const half_t* F = ff + img * 16384;                   // NHWC
  float* O = lnf + img * 16384;
  for (int i = threadIdx.x; i < 16384; i += 256) {
    int c = i >> 8, p = i & 255;
    O[i] = (A[i] - m) * inv * lng[i] + lnb[i] + (float)F[p * 64 + c];
  }
}

// CBAM + final embed combine:  e = lnf + sigmoid(sam)*y + 2*f   (e CHW f32)
__global__ void __launch_bounds__(256) k_cbam(
    const half_t* ff, const float* lnf,
    const float* caW1, const float* cab1, const float* caW2, const float* cab2,
    const float* saW, float* e) {
  __shared__ half_t yv[64 * 256];
  __shared__ float chmax[64], chavg[64], gch[64], hbuf[8], smx[256], sav[256];
  int img = blockIdx.x, tid = threadIdx.x;
  const half_t* X = ff + img * 16384;                   // NHWC [p][c]
  if (tid < 64) {
    float mx = -1e30f, s = 0.f;
    for (int p = 0; p < 256; ++p) {
      float v = (float)X[p * 64 + tid];
      mx = fmaxf(mx, v); s += v;
    }
    chmax[tid] = mx; chavg[tid] = s * (1.f / 256.f);
  }
  __syncthreads();
  if (tid < 8) {
    int j = tid & 3;
    const float* v = (tid < 4) ? chmax : chavg;
    float h = cab1[j];
    for (int c = 0; c < 64; ++c) h += caW1[j * 64 + c] * v[c];
    hbuf[tid] = fmaxf(h, 0.f);
  }
  __syncthreads();
  if (tid < 64) {
    float o = 2.f * cab2[tid];
    for (int j = 0; j < 4; ++j) o += caW2[tid * 4 + j] * (hbuf[j] + hbuf[4 + j]);
    gch[tid] = sigmoidf_(o);
  }
  __syncthreads();
  for (int i = tid; i < 16384; i += 256) {
    int c = i >> 8, p = i & 255;
    yv[i] = (half_t)(gch[c] * (float)X[p * 64 + c]);    // yv is [c][p]
  }
  __syncthreads();
  {
    float mx = -1e30f, s = 0.f;
    for (int c = 0; c < 64; ++c) {
      float v = (float)yv[c * 256 + tid];
      mx = fmaxf(mx, v); s += v;
    }
    smx[tid] = mx; sav[tid] = s * (1.f / 64.f);
  }
  __syncthreads();
  {
    int py = tid >> 4, px = tid & 15;
    float s = 0.f;
    for (int u = 0; u < 7; ++u)
      for (int v = 0; v < 7; ++v) {
        int yy = py + u - 3, xx = px + v - 3;
        if (yy >= 0 && yy < 16 && xx >= 0 && xx < 16) {
          int pp = yy * 16 + xx;
          s += saW[u * 7 + v] * smx[pp] + saW[49 + u * 7 + v] * sav[pp];
        }
      }
    float sg = sigmoidf_(s);
    float* E = e + img * 16384;
    const float* L = lnf + img * 16384;
    for (int c = 0; c < 64; ++c) {
      int idx = c * 256 + tid;
      E[idx] = L[idx] + sg * (float)yv[idx] + 2.f * (float)X[tid * 64 + c];
    }
  }
}

// ---------------------------------------------------------------- cov + sim
__global__ void __launch_bounds__(256) k_cov_mean(const float* e, float* covmu) {
  int w = blockIdx.x >> 6, c = blockIdx.x & 63;
  float s = 0.f;
  for (int i = threadIdx.x; i < 2560; i += 256) {
    int sh = i >> 8, p = i & 255;
    s += e[((128 + w * 10 + sh) * 64 + c) * 256 + p];
  }
  __shared__ float bs[256];
  bs[threadIdx.x] = s; __syncthreads();
  for (int k = 128; k > 0; k >>= 1) {
    if (threadIdx.x < k) bs[threadIdx.x] += bs[threadIdx.x + k];
    __syncthreads();
  }
  if (threadIdx.x == 0) covmu[w * 64 + c] = bs[0] / 2560.f;
}

__global__ void __launch_bounds__(256) k_cov(const float* e, const float* covmu,
                                             half_t* covh) {
  int w = blockIdx.x / 16;
  int pair = (blockIdx.x % 16) * 256 + threadIdx.x;   // c*64 + d
  int c = pair >> 6, d = pair & 63;
  float mc = covmu[w * 64 + c], md = covmu[w * 64 + d];
  float acc = 0.f;
  for (int sh = 0; sh < 10; ++sh) {
    const float* ec = e + ((128 + w * 10 + sh) * 64 + c) * 256;
    const float* ed = e + ((128 + w * 10 + sh) * 64 + d) * 256;
    for (int p = 0; p < 256; ++p) acc += (ec[p] - mc) * (ed[p] - md);
  }
  covh[w * 4096 + pair] = (half_t)(acc / 2559.f);
}

// row-normalize query features; output NHWC f16 [b][p][c]
__global__ void __launch_bounds__(256) k_qnorm(const float* e, half_t* Qn) {
  int row = blockIdx.x;                    // b*64 + c, b<128
  float v = e[row * 256 + threadIdx.x];
  __shared__ float bs[256];
  bs[threadIdx.x] = v * v; __syncthreads();
  for (int k = 128; k > 0; k >>= 1) {
    if (threadIdx.x < k) bs[threadIdx.x] += bs[threadIdx.x + k];
    __syncthreads();
  }
  int b = row >> 6, c = row & 63;
  Qn[b * 16384 + threadIdx.x * 64 + c] = (half_t)(v * rsqrtf(bs[0] + 1e-20f));
}

// sim[b,w,p] = diag(Q^T Cov Q); fused leaky + clsW reduction -> out[b,w]
__global__ void __launch_bounds__(256) k_sim(const half_t* Qn, const half_t* covh,
                                             const float* clsW, float* outp) {
  __shared__ float simb[256];
  __shared__ float red2[256];
  int b = blockIdx.x / 5, w = blockIdx.x % 5;
  int tid = threadIdx.x, lane = tid & 31, wv = tid >> 5;
  bool hiL = (lane & 16) != 0;
  int kbA = hiL ? 8 : 0, koB = hiL ? 8 : 0;
  const half_t* Qb = Qn + b * 16384;                   // NHWC [p][c]
  const half_t* C  = covh + w * 4096;
  for (int pt = wv; pt < 16; pt += 8) {
    int col = pt * 16 + (lane & 15);
    // separate accumulator sets per K-chunk -> all 8 WMMAs independent
    v8f s0 = {}, s1 = {}, s2 = {}, s3 = {};
    v8f t0 = {}, t1 = {}, t2 = {}, t3 = {};
    {
      v16h bf = fragB_kcontig(Qb + col * 64 + koB);
      const half_t* crow = C + kbA;
      s0 = wmma16(frag_kcontig(crow + ((lane & 15) + 0)  * 64), bf, s0);
      s1 = wmma16(frag_kcontig(crow + ((lane & 15) + 16) * 64), bf, s1);
      s2 = wmma16(frag_kcontig(crow + ((lane & 15) + 32) * 64), bf, s2);
      s3 = wmma16(frag_kcontig(crow + ((lane & 15) + 48) * 64), bf, s3);
    }
    {
      v16h bf = fragB_kcontig(Qb + col * 64 + 32 + koB);
      const half_t* crow = C + 32 + kbA;
      t0 = wmma16(frag_kcontig(crow + ((lane & 15) + 0)  * 64), bf, t0);
      t1 = wmma16(frag_kcontig(crow + ((lane & 15) + 16) * 64), bf, t1);
      t2 = wmma16(frag_kcontig(crow + ((lane & 15) + 32) * 64), bf, t2);
      t3 = wmma16(frag_kcontig(crow + ((lane & 15) + 48) * 64), bf, t3);
    }
    int rbase = hiL ? 8 : 0;
    v8h q0 = *(const v8h*)(Qb + col * 64 + 0  + rbase);
    v8h q1 = *(const v8h*)(Qb + col * 64 + 16 + rbase);
    v8h q2 = *(const v8h*)(Qb + col * 64 + 32 + rbase);
    v8h q3 = *(const v8h*)(Qb + col * 64 + 48 + rbase);
    float part = 0.f;
#pragma unroll
    for (int r = 0; r < 8; ++r) {
      part += (s0[r] + t0[r]) * (float)q0[r];
      part += (s1[r] + t1[r]) * (float)q1[r];
      part += (s2[r] + t2[r]) * (float)q2[r];
      part += (s3[r] + t3[r]) * (float)q3[r];
    }
    part += __shfl_xor(part, 16);   // combine the two row-halves (wave32)
    if (lane < 16) simb[col] = part;
  }
  __syncthreads();
  red2[tid] = leaky(simb[tid]) * clsW[tid];
  __syncthreads();
  for (int k = 128; k > 0; k >>= 1) {
    if (tid < k) red2[tid] += red2[tid + k];
    __syncthreads();
  }
  if (tid == 0) {
    float t = red2[0];
    outp[b * 5 + w]        = t;
    outp[640 + b * 5 + w]  = t;
    outp[1280 + b * 5 + w] = t;
  }
}

// ---------------------------------------------------------------- launch
extern "C" void kernel_launch(void* const* d_in, const int* in_sizes, int n_in,
                              void* d_out, int out_size, void* d_ws, size_t ws_size,
                              hipStream_t stream) {
  (void)in_sizes; (void)n_in; (void)out_size; (void)ws_size;
  const float* in1  = (const float*)d_in[0];
  const float* in2  = (const float*)d_in[1];
  const float* fW0  = (const float*)d_in[2];
  const float* fW1  = (const float*)d_in[3];
  const float* fW2  = (const float*)d_in[4];
  const float* fW3  = (const float*)d_in[5];
  const float* fg0 = (const float*)d_in[6],  *fb0 = (const float*)d_in[7];
  const float* fg1 = (const float*)d_in[8],  *fb1 = (const float*)d_in[9];
  const float* fg2 = (const float*)d_in[10], *fb2 = (const float*)d_in[11];
  const float* fg3 = (const float*)d_in[12], *fb3 = (const float*)d_in[13];
  const float* Wq = (const float*)d_in[14]; const float* bq = (const float*)d_in[15];
  const float* Wk = (const float*)d_in[16]; const float* bk = (const float*)d_in[17];
  const float* Wv = (const float*)d_in[18]; const float* bv = (const float*)d_in[19];
  const float* rel_h = (const float*)d_in[20];
  const float* rel_w = (const float*)d_in[21];
  const float* ln_g  = (const float*)d_in[22];
  const float* ln_b  = (const float*)d_in[23];
  const float* caW1 = (const float*)d_in[24]; const float* cab1 = (const float*)d_in[25];
  const float* caW2 = (const float*)d_in[26]; const float* cab2 = (const float*)d_in[27];
  const float* saW  = (const float*)d_in[28];
  const float* clsW = (const float*)d_in[29];
  float* outp = (float*)d_out;

  uintptr_t base = (uintptr_t)d_ws;
  size_t cur = 0;
  auto alloc = [&](size_t bytes) -> void* {
    cur = (cur + 255) & ~(size_t)255;
    void* p = (void*)(base + cur);
    cur += bytes;
    return p;
  };
  half_t* Wt1  = (half_t*)alloc(36864 * 2);
  half_t* Wt2  = (half_t*)alloc(36864 * 2);
  half_t* Wt3  = (half_t*)alloc(36864 * 2);
  half_t* Wqh  = (half_t*)alloc(4096 * 2);
  half_t* Wkh  = (half_t*)alloc(4096 * 2);
  half_t* Wvh  = (half_t*)alloc(4096 * 2);
  float*  part = (float*)alloc((size_t)NIMG * 64 * 2 * 4);
  float*  mi   = (float*)alloc(384 * 2 * 4);
  half_t* x1 = (half_t*)alloc((size_t)NIMG * 64 * 1024 * 2);  // reused by attn+lnf
  half_t* y1 = (half_t*)alloc((size_t)NIMG * 64 * 1024 * 2);  // reused by e
  half_t* x2 = (half_t*)alloc((size_t)NIMG * 64 * 256 * 2);
  half_t* y2 = (half_t*)alloc((size_t)NIMG * 64 * 256 * 2);
  half_t* x3 = (half_t*)alloc((size_t)NIMG * 64 * 256 * 2);
  half_t* y3 = (half_t*)alloc((size_t)NIMG * 64 * 256 * 2);
  half_t* ff = (half_t*)alloc((size_t)NIMG * 64 * 256 * 2);
  half_t* Qn = (half_t*)alloc((size_t)128 * 64 * 256 * 2);
  float*  covmu = (float*)alloc(5 * 64 * 4);
  half_t* covh  = (half_t*)alloc(5 * 4096 * 2);
  // aliases (features are dead before these are written)
  float* attn = (float*)x1;                                    // 11.66 MB
  float* lnf  = (float*)((uintptr_t)x1 + (size_t)NIMG * 64 * 256 * 4);
  float* e    = (float*)y1;                                    // 11.66 MB

  // ---- features ----
  k_prep<<<144, 256, 0, stream>>>(fW1, fW2, fW3, Wq, Wk, Wv,
                                  Wt1, Wt2, Wt3, Wqh, Wkh, Wvh);
  k_conv0_stats<<<NIMG * 64, 256, 0, stream>>>(in1, in2, fW0, part);
  k_finalize_stats<<<1, 384, 0, stream>>>(part, mi, 4096);
  k_conv0_apply<<<NIMG * 64 * 1024 / 256, 256, 0, stream>>>(in1, in2, fW0, mi,
                                                            fg0, fb0, x1);
  k_conv_gemm<32, 32><<<NIMG * 8, 256, 0, stream>>>(x1, Wt1, y1);
  k_bn_stats<<<NIMG * 64, 256, 0, stream>>>(y1, part, 1024);
  k_finalize_stats<<<1, 384, 0, stream>>>(part, mi, 1024);
  k_bn_apply<<<NIMG * 64 * 256 / 256, 256, 0, stream>>>(y1, x2, mi, fg1, fb1, 32, 32, 1);
  k_conv_gemm<16, 16><<<NIMG * 2, 256, 0, stream>>>(x2, Wt2, y2);
  k_bn_stats<<<NIMG * 64, 256, 0, stream>>>(y2, part, 256);
  k_finalize_stats<<<1, 384, 0, stream>>>(part, mi, 256);
  k_bn_apply<<<NIMG * 64 * 256 / 256, 256, 0, stream>>>(y2, x3, mi, fg2, fb2, 16, 16, 0);
  k_conv_gemm<16, 16><<<NIMG * 2, 256, 0, stream>>>(x3, Wt3, y3);
  k_bn_stats<<<NIMG * 64, 256, 0, stream>>>(y3, part, 256);
  k_finalize_stats<<<1, 384, 0, stream>>>(part, mi, 256);
  k_bn_apply<<<NIMG * 64 * 256 / 256, 256, 0, stream>>>(y3, ff, mi, fg3, fb3, 16, 16, 0);
  // ---- embed ----
  k_mhsa<<<NIMG * 4, 256, 0, stream>>>(ff, Wqh, Wkh, Wvh, bq, bk, bv,
                                       rel_h, rel_w, attn);
  k_ln<<<NIMG, 256, 0, stream>>>(attn, ff, ln_g, ln_b, lnf);
  k_cbam<<<NIMG, 256, 0, stream>>>(ff, lnf, caW1, cab1, caW2, cab2, saW, e);
  // ---- covariance metric + classification ----
  k_cov_mean<<<5 * 64, 256, 0, stream>>>(e, covmu);
  k_cov<<<80, 256, 0, stream>>>(e, covmu, covh);
  k_qnorm<<<128 * 64, 256, 0, stream>>>(e, Qn);
  k_sim<<<640, 256, 0, stream>>>(Qn, covh, clsW, outp);
}
